// GraphTransformerLayer_46514495816136
// MI455X (gfx1250) — matrix operand order repaired and checked by reading, not compiled, not run on hardware
//
#include <hip/hip_runtime.h>
#include <hip/hip_fp16.h>
#include <math.h>

#define NN 20000          // nodes
#define NE 320000         // edges
#define DM 256            // model dim (= H*C)
#define NH 4              // heads
// concat projection output dim = 4*DM = 1024

typedef __attribute__((ext_vector_type(16))) _Float16 v16h;
typedef __attribute__((ext_vector_type(8)))  _Float16 v8h;
typedef __attribute__((ext_vector_type(8)))  float    v8f;

union HFrag { v16h v; v8h h[2]; };

// ---------------------------------------------------------------- converts
__global__ void cvt_x_f16(const float* __restrict__ x, _Float16* __restrict__ xh) {
  size_t i = (size_t)blockIdx.x * 256 + threadIdx.x;   // grid covers NN*DM
  xh[i] = (_Float16)x[i];
}

// wt[n][k] = Wconcat[k][n], n in [0,1024), k in [0,256), f16
__global__ void cvt_w_f16(const float* __restrict__ Wq, const float* __restrict__ Wk,
                          const float* __restrict__ Wv, const float* __restrict__ Ws,
                          _Float16* __restrict__ wt) {
  int i = blockIdx.x * 256 + threadIdx.x;              // i = n*256 + k, i < 1024*256
  int n = i >> 8;
  int k = i & 255;
  int sel = n >> 8;
  int c   = n & 255;
  const float* W = (sel == 0) ? Wq : (sel == 1) ? Wk : (sel == 2) ? Wv : Ws;
  wt[i] = (_Float16)W[(size_t)k * 256 + c];
}

__global__ void init_minf(float* __restrict__ m, int n) {
  int i = blockIdx.x * 256 + threadIdx.x;
  if (i < n) m[i] = -INFINITY;
}

// ---------------------------------------------------------------- WMMA GEMM
// Wave computes a 16x64 output slab (4 accumulators) so the A fragment is
// loaded once per K-step and reused across 4 WMMAs.
__global__ void __launch_bounds__(256) gemm_qkvs(
    const _Float16* __restrict__ xh, const _Float16* __restrict__ wt,
    const float* __restrict__ bq, const float* __restrict__ bk,
    const float* __restrict__ bv, const float* __restrict__ bs,
    float* __restrict__ qb, float* __restrict__ kb,
    float* __restrict__ vb, float* __restrict__ sb) {
  const int lane = threadIdx.x & 31;
  const int wid  = threadIdx.x >> 5;
  const int mt   = blockIdx.x;                 // 0..1249  (16-row slab)
  const int ng   = blockIdx.y * 8 + wid;       // 0..15    (64-col group)
  const int hi   = lane >> 4;                  // half-wave select
  const int mn   = lane & 15;                  // A row / B col within tile

  const _Float16* arow = xh + (size_t)(mt * 16 + mn) * 256;
  const _Float16* brow = wt + (size_t)(ng * 64 + mn) * 256;  // +j*16*256 per n-tile
  const int ka  = hi * 8;    // A K offset for hi lanes
  const int kb2 = hi * 8;    // B K offset for hi lanes

  v8f acc[4] = {};
#pragma unroll
  for (int kk = 0; kk < 256; kk += 32) {
    HFrag a;
    a.h[0] = *(const v8h*)(arow + kk + ka);
    a.h[1] = *(const v8h*)(arow + kk + 16 + ka);
#pragma unroll
    for (int j = 0; j < 4; ++j) {
      HFrag b;
      const _Float16* bj = brow + (size_t)j * 16 * 256 + kk + kb2;
      b.h[0] = *(const v8h*)(bj);
      b.h[1] = *(const v8h*)(bj + 16);
      acc[j] = __builtin_amdgcn_wmma_f32_16x16x32_f16(false, a.v, false, b.v,
                                                      (short)0, acc[j], false, false);
    }
  }

  // 64-col group never straddles a 256-col boundary -> uniform buffer select
  const int buf = ng >> 2;
  const float* bp = (buf == 0) ? bq : (buf == 1) ? bk : (buf == 2) ? bv : bs;
  float*       op = (buf == 0) ? qb : (buf == 1) ? kb : (buf == 2) ? vb : sb;
#pragma unroll
  for (int j = 0; j < 4; ++j) {
    const int c = (ng * 64 + j * 16 + mn) & 255;   // column within buffer
    const float bias = bp[c];
#pragma unroll
    for (int r = 0; r < 8; ++r) {
      const int row = mt * 16 + hi * 8 + r;        // D layout: VGPR r -> M = r + 8*hi
      op[(size_t)row * 256 + c] = acc[j][r] + bias;
    }
  }
}

// ---------------------------------------------------------------- edge logits
__device__ __forceinline__ void atomicMaxFloat(float* addr, float v) {
  if (v >= 0.0f) atomicMax((int*)addr, __float_as_int(v));
  else           atomicMin((unsigned int*)addr, __float_as_uint(v));
}

// wave per edge; lane l covers channels [l*8, l*8+8), head = l>>3
__global__ void __launch_bounds__(256) edge_logits(
    const int* __restrict__ ei, const float* __restrict__ qb,
    const float* __restrict__ kb, float* __restrict__ alpha,
    float* __restrict__ mmax) {
  const int e    = blockIdx.x * 8 + (threadIdx.x >> 5);
  const int lane = threadIdx.x & 31;
  const int h    = lane >> 3;
  const int sub  = lane & 7;
  const int src  = ei[e];
  const int dst  = ei[NE + e];
  const float4* qp = (const float4*)(qb + (size_t)dst * 256 + lane * 8);
  const float4* kp = (const float4*)(kb + (size_t)src * 256 + lane * 8);
  float4 q0 = qp[0], q1 = qp[1];
  float4 k0 = kp[0], k1 = kp[1];
  float s = q0.x * k0.x + q0.y * k0.y + q0.z * k0.z + q0.w * k0.w +
            q1.x * k1.x + q1.y * k1.y + q1.z * k1.z + q1.w * k1.w;
  s += __shfl_xor(s, 1);
  s += __shfl_xor(s, 2);
  s += __shfl_xor(s, 4);
  if (sub == 0) {
    const float logit = s * 0.125f;            // 1/sqrt(64)
    alpha[(size_t)e * NH + h] = logit;
    atomicMaxFloat(mmax + (size_t)dst * NH + h, logit);
  }
}

// thread per (edge, head): a = exp(alpha - m[dst]); denom[dst] += a
__global__ void __launch_bounds__(256) edge_expsum(
    const int* __restrict__ ei, float* __restrict__ alpha,
    const float* __restrict__ mmax, float* __restrict__ denom) {
  const int i = blockIdx.x * 256 + threadIdx.x;   // i < NE*NH
  const int e = i >> 2;
  const int h = i & 3;
  const int dst = ei[NE + e];
  const float a = __expf(alpha[i] - mmax[(size_t)dst * NH + h]);
  alpha[i] = a;
  atomicAdd(denom + (size_t)dst * NH + h, a);
}

// wave per edge: agg[dst] += a_unnorm * v[src]  (division folded into finalize)
__global__ void __launch_bounds__(256) edge_scatter(
    const int* __restrict__ ei, const float* __restrict__ alpha,
    const float* __restrict__ vb, float* __restrict__ agg) {
  const int e    = blockIdx.x * 8 + (threadIdx.x >> 5);
  const int lane = threadIdx.x & 31;
  const int h    = lane >> 3;
  const int src  = ei[e];
  const int dst  = ei[NE + e];
  const float a  = alpha[(size_t)e * NH + h];
  const float* vp = vb  + (size_t)src * 256 + lane * 8;
  float*       ap = agg + (size_t)dst * 256 + lane * 8;
#pragma unroll
  for (int t = 0; t < 8; ++t) atomicAdd(ap + t, a * vp[t]);
}

// ---------------------------------------------------------------- finalize
// wave per node: agg/denom + skip -> GELU(exact erf) -> LayerNorm -> out
__global__ void __launch_bounds__(256) finalize(
    const float* __restrict__ agg, const float* __restrict__ denom,
    const float* __restrict__ skip, const float* __restrict__ gamma,
    const float* __restrict__ beta, float* __restrict__ out) {
  const int n    = blockIdx.x * 8 + (threadIdx.x >> 5);
  const int lane = threadIdx.x & 31;
  const int h    = lane >> 3;
  const float den = denom[(size_t)n * NH + h] + 1e-16f;
  const float rden = 1.0f / den;
  const float* ag = agg  + (size_t)n * 256 + lane * 8;
  const float* sk = skip + (size_t)n * 256 + lane * 8;
  float vals[8];
  float s = 0.0f, ss = 0.0f;
#pragma unroll
  for (int t = 0; t < 8; ++t) {
    const float x = ag[t] * rden + sk[t];
    const float g = 0.5f * x * (1.0f + erff(x * 0.70710678118654752440f));
    vals[t] = g;
    s += g;
    ss += g * g;
  }
#pragma unroll
  for (int m = 1; m < 32; m <<= 1) {
    s  += __shfl_xor(s, m);
    ss += __shfl_xor(ss, m);
  }
  const float mu   = s * (1.0f / 256.0f);
  const float var  = ss * (1.0f / 256.0f) - mu * mu;
  const float rstd = rsqrtf(var + 1e-5f);
  float* op = out + (size_t)n * 256 + lane * 8;
#pragma unroll
  for (int t = 0; t < 8; ++t) {
    const int d = lane * 8 + t;
    op[t] = (vals[t] - mu) * rstd * gamma[d] + beta[d];
  }
}

// ---------------------------------------------------------------- launch
extern "C" void kernel_launch(void* const* d_in, const int* in_sizes, int n_in,
                              void* d_out, int out_size, void* d_ws, size_t ws_size,
                              hipStream_t stream) {
  const float* x    = (const float*)d_in[0];
  const int*   ei   = (const int*)d_in[1];
  const float* Wq   = (const float*)d_in[2];
  const float* bq   = (const float*)d_in[3];
  const float* Wk   = (const float*)d_in[4];
  const float* bk   = (const float*)d_in[5];
  const float* Wv   = (const float*)d_in[6];
  const float* bv   = (const float*)d_in[7];
  const float* Ws   = (const float*)d_in[8];
  const float* bs   = (const float*)d_in[9];
  const float* gamma = (const float*)d_in[10];
  const float* beta  = (const float*)d_in[11];
  float* out = (float*)d_out;

  char* ws = (char*)d_ws;
  const size_t SZ_XH  = (size_t)NN * DM * 2;       // 10,240,000
  const size_t SZ_WT  = (size_t)1024 * 256 * 2;    //    524,288
  const size_t SZ_F   = (size_t)NN * DM * 4;       // 20,480,000
  const size_t SZ_NH4 = (size_t)NN * NH * 4;       //    320,000
  const size_t SZ_AL  = (size_t)NE * NH * 4;       //  5,120,000

  size_t off = 0;
  _Float16* xh   = (_Float16*)(ws + off); off += SZ_XH;
  _Float16* wt   = (_Float16*)(ws + off); off += SZ_WT;
  float*    qb   = (float*)(ws + off);    off += SZ_F;
  float*    kb   = (float*)(ws + off);    off += SZ_F;
  float*    vb   = (float*)(ws + off);    off += SZ_F;
  float*    sb   = (float*)(ws + off);    off += SZ_F;
  float*    agg  = (float*)(ws + off);    off += SZ_F;
  float*    den  = (float*)(ws + off);    off += SZ_NH4;   // contiguous with agg
  float*    alph = (float*)(ws + off);    off += SZ_AL;
  float*    mmax = (float*)(ws + off);    off += SZ_NH4;

  // zero agg + denom (contiguous); -inf seed for segment max
  hipMemsetAsync(agg, 0, SZ_F + SZ_NH4, stream);
  init_minf<<<(NN * NH + 255) / 256, 256, 0, stream>>>(mmax, NN * NH);

  cvt_x_f16<<<(NN * DM) / 256, 256, 0, stream>>>(x, xh);
  cvt_w_f16<<<(1024 * 256) / 256, 256, 0, stream>>>(Wq, Wk, Wv, Ws, wt);

  // 16 waves of 64 cols cover 1024 output cols; 2 blocks (8 waves each) in y
  gemm_qkvs<<<dim3(NN / 16, 2), 256, 0, stream>>>(xh, wt, bq, bk, bv, bs,
                                                  qb, kb, vb, sb);

  edge_logits<<<NE / 8, 256, 0, stream>>>(ei, qb, kb, alph, mmax);
  edge_expsum<<<(NE * NH) / 256, 256, 0, stream>>>(ei, alph, mmax, den);
  edge_scatter<<<NE / 8, 256, 0, stream>>>(ei, alph, vb, agg);

  finalize<<<NN / 8, 256, 0, stream>>>(agg, den, sb, gamma, beta, out);
}
